// S4BlockWrapper_48223892799529
// MI455X (gfx1250) — matrix operand order repaired and checked by reading, not compiled, not run on hardware
//
#include <hip/hip_runtime.h>
#include <hip/hip_bf16.h>

// ---------------------------------------------------------------------------
// S4 block for MI455X (gfx1250, wave32, WMMA).
//   B=16, L=1024, H=1024, N=64
// Pipeline:
//   K1: kern[h,l] = sum_n Re(2*coef * exp(dtA*l))            (VALU/trans)
//   K2: xt[h][b][l] = x[b][l][h]                             (LDS transpose)
//   K3: y = gelu(conv_causal(x,kern) + x*D) -> bf16          (v_wmma_f32_16x16x4_f32,
//                                                             global_load_async_to_lds_b128)
//   K4: z = y @ W + b ; out = glu(z)                         (v_wmma_f32_16x16x32_bf16,
//                                                             double-buffered LDS)
// ---------------------------------------------------------------------------

typedef __attribute__((ext_vector_type(2)))  float   v2f;
typedef __attribute__((ext_vector_type(4)))  float   v4f;
typedef __attribute__((ext_vector_type(8)))  float   v8f;
typedef __attribute__((ext_vector_type(8)))  __bf16  v8bf;
typedef __attribute__((ext_vector_type(16))) __bf16  v16bf;

#define S4_B   16
#define S4_L   1024
#define S4_H   1024
#define S4_N   64
#define S4_H2  2048

__device__ __forceinline__ float s4_gelu(float y) {
    // tanh approximation (jax.nn.gelu default)
    float y3 = y * y * y;
    return 0.5f * y * (1.f + tanhf(0.7978845608028654f * (y + 0.044715f * y3)));
}
__device__ __forceinline__ float s4_sigmoid(float v) {
    return 1.f / (1.f + __expf(-v));
}

// ---------------------------------------------------------------------------
// K1: one block per h. Threads 0..63 compute per-n coefficients into LDS,
// then all 256 threads evaluate kern[h, l] for 4 l's each.
// ---------------------------------------------------------------------------
__global__ __launch_bounds__(256) void s4_kern_gen(
    const float* __restrict__ log_dt, const float* __restrict__ arl,
    const float* __restrict__ aim,    const float* __restrict__ cre,
    const float* __restrict__ cim,    float* __restrict__ kern) {
    __shared__ v4f lc[S4_N];
    const int h = blockIdx.x;
    const int t = threadIdx.x;
    if (t < S4_N) {
        float dt  = __expf(log_dt[h]);
        float Ar  = -__expf(arl[h * S4_N + t]);
        float Ai  = aim[h * S4_N + t];
        float are = dt * Ar, aimv = dt * Ai;
        float er  = __expf(are);
        float s, c;
        __sincosf(aimv, &s, &c);
        float emr = er * c - 1.f;        // Re(exp(dtA) - 1)
        float emi = er * s;              // Im(exp(dtA) - 1)
        float Cr  = cre[h * S4_N + t], Ci = cim[h * S4_N + t];
        float nre = Cr * emr - Ci * emi;
        float nim = Cr * emi + Ci * emr;
        float den = 1.f / (Ar * Ar + Ai * Ai);
        v4f v;
        v.x = are;
        v.y = aimv;
        v.z = 2.f * (nre * Ar + nim * Ai) * den;   // 2*Re(coef)
        v.w = 2.f * (nim * Ar - nre * Ai) * den;   // 2*Im(coef)
        lc[t] = v;
    }
    __syncthreads();
#pragma unroll
    for (int q = 0; q < 4; ++q) {
        int   l  = t + q * 256;
        float fl = (float)l;
        float acc = 0.f;
        for (int n = 0; n < S4_N; ++n) {
            v4f v = lc[n];
            float e = __expf(v.x * fl);
            float s, c;
            __sincosf(v.y * fl, &s, &c);
            acc += e * (v.z * c - v.w * s);
        }
        kern[h * S4_L + l] = acc;
    }
}

// ---------------------------------------------------------------------------
// K2: transpose x[b][l][h] -> xt[h][b][l].  32x32 LDS tiles, block 32x8.
// ---------------------------------------------------------------------------
__global__ __launch_bounds__(256) void s4_transpose(
    const float* __restrict__ x, float* __restrict__ xt) {
    __shared__ float tile[32][33];
    const int b  = blockIdx.z;
    const int l0 = blockIdx.y * 32;
    const int h0 = blockIdx.x * 32;
    const int tx = threadIdx.x, ty = threadIdx.y;
#pragma unroll
    for (int q = 0; q < 4; ++q) {
        int l = l0 + ty + q * 8;
        tile[ty + q * 8][tx] = x[(b * S4_L + l) * S4_H + h0 + tx];
    }
    __syncthreads();
#pragma unroll
    for (int q = 0; q < 4; ++q) {
        int h = h0 + ty + q * 8;
        xt[(h * S4_B + b) * S4_L + l0 + tx] = tile[tx][ty + q * 8];
    }
}

// ---------------------------------------------------------------------------
// K3: causal Toeplitz conv via v_wmma_f32_16x16x4_f32.
// One block per h (8 waves). Kern row (zero-padded by 16) staged into LDS
// with the gfx1250 async global->LDS copy (ASYNCcnt-tracked).
// Output tile for wave: D[b=0..15][l=l0..l0+15]; K loop over time j in steps
// of 4: A = xt[h][b][j..j+3] (16x4), B[k][n] = kern[l0+n-j0-k] (4x16).
// Epilogue: gelu(y + x*D[h]) -> yt (bf16, [h][b][l]).
// ---------------------------------------------------------------------------
__global__ __launch_bounds__(256) void s4_conv_wmma(
    const float* __restrict__ xt, const float* __restrict__ kern,
    const float* __restrict__ Dp, __bf16* __restrict__ yt) {
    __shared__ float lk[16 + S4_L];   // lk[16+i] = kern[h][i], lk[0..15] = 0
    const int h = blockIdx.x;
    const int t = threadIdx.x;

    if (t < 16) lk[t] = 0.f;
    {
        // async copy: 256 threads x 16B = 4KB kern row -> LDS.
        // flat LDS address low 32 bits == LDS byte offset (aperture rule).
        unsigned ldsoff = (unsigned)(unsigned long long)(lk + 16 + t * 4);
        unsigned long long gaddr =
            (unsigned long long)(const void*)(kern + h * S4_L + t * 4);
        asm volatile("global_load_async_to_lds_b128 %0, %1, off"
                     :: "v"(ldsoff), "v"(gaddr) : "memory");
        asm volatile("s_wait_asynccnt 0x0" ::: "memory");
    }
    __syncthreads();

    const int   wave = t >> 5, lane = t & 31;
    const int   half = lane >> 4;          // k-half for A/B fragments
    const int   bl   = lane & 15;          // m (=batch) for A, n (=l-col) for B
    const float Dh   = Dp[h];
    const float* xrow = xt + (h * S4_B + bl) * S4_L;   // A source row (b = bl)

    for (int lt = wave; lt < S4_L / 16; lt += 8) {
        const int l0 = lt * 16;
        v8f acc = {0.f, 0.f, 0.f, 0.f, 0.f, 0.f, 0.f, 0.f};
        for (int j0 = 0; j0 <= l0 + 12; j0 += 4) {
            v2f a = *(const v2f*)(xrow + j0 + half * 2);     // K = half*2, half*2+1
            int d = l0 + bl - j0 - half * 2;                 // kern offset for k=half*2
            v2f bfr;
            bfr.x = lk[16 + d];                              // B[half*2    ][n]
            bfr.y = lk[15 + d];                              // B[half*2 + 1][n]
            acc = __builtin_amdgcn_wmma_f32_16x16x4_f32(
                false, a, false, bfr, (short)0, acc, false, false);
        }
        // epilogue: D rows M = r + half*8, cols N = bl
#pragma unroll
        for (int r = 0; r < 8; ++r) {
            int   brow = r + half * 8;
            int   l    = l0 + bl;
            float xv   = xt[(h * S4_B + brow) * S4_L + l];
            float g    = s4_gelu(acc[r] + xv * Dh);
            yt[(h * S4_B + brow) * S4_L + l] = (__bf16)g;
        }
    }
}

// ---------------------------------------------------------------------------
// K4: Z = Y @ W + b, out = glu(Z) via v_wmma_f32_16x16x32_bf16.
// Block = 256 thr (8 waves) computes a 64(m) x 64(n) output tile; waves are
// split (m_sub 0..3) x (a-half / g-half).  A/B tiles double-buffered in LDS
// (global loads for tile kt+1 overlap the WMMAs on tile kt); A and B
// use contiguous k order (k = 16*half + s): each lane's fragment is one 32B LDS read.
// ---------------------------------------------------------------------------
__global__ __launch_bounds__(256) void s4_gemm_glu(
    const __bf16* __restrict__ yt, const float* __restrict__ W,
    const float* __restrict__ bout, float* __restrict__ out) {
    __shared__ __align__(16) __bf16 As[2][64 * 40];    // [m][k]
    __shared__ __align__(16) __bf16 Bs[2][128 * 40];   // [n_local][k]; n<64: a, >=64: g
    __shared__ __align__(16) float  Zs[64 * 132];

    const int t    = threadIdx.x;
    const int m0   = blockIdx.x * 64;
    const int n0   = blockIdx.y * 64;
    const int bidx = m0 >> 10;          // batch (m-tile never crosses batch)
    const int l0   = m0 & (S4_L - 1);

    const int wave = t >> 5, lane = t & 31;
    const int msub = wave & 3, hsel = wave >> 2;    // hsel: 0 = a-half, 1 = g-half
    const int ln15 = lane & 15, lh = lane >> 4;

    // A tile loader: thread -> (k_local, 8 contiguous m)
    const int a_hl  = t >> 3;              // 0..31
    const int a_mo  = (t & 7) * 8;         // 0..56
    const int kperm = a_hl;
    // B tile loader: thread -> (2 n cols, k base)
    const int b_nl = (t & 63) * 2;         // 0..126
    const int b_kb = t >> 6;               // 0..3
    const int col0 = n0 + (b_nl & 63) + ((b_nl >> 6) << 10);

    const __bf16* aSrc = yt + (a_hl * S4_B + bidx) * S4_L + l0 + a_mo;
    const float*  bSrc = W + b_kb * S4_H2 + col0;
    const size_t  aStep = (size_t)32 * S4_B * S4_L;   // elems per K-tile
    const size_t  bStep = (size_t)32 * S4_H2;

    const v8f vz = {0.f, 0.f, 0.f, 0.f, 0.f, 0.f, 0.f, 0.f};
    v8f acc[4] = {vz, vz, vz, vz};

    auto stage = [&](int buf, v8bf aR, const v2f* bR) {
#pragma unroll
        for (int q = 0; q < 8; ++q) As[buf][(a_mo + q) * 40 + kperm] = aR[q];
#pragma unroll
        for (int p = 0; p < 8; ++p) {
            int kk = b_kb + p * 4;
            Bs[buf][b_nl * 40 + kk]       = (__bf16)bR[p].x;
            Bs[buf][(b_nl + 1) * 40 + kk] = (__bf16)bR[p].y;
        }
    };

    // prologue: stage tile 0
    {
        v8bf aR = *(const v8bf*)aSrc;
        v2f  bR[8];
#pragma unroll
        for (int p = 0; p < 8; ++p) bR[p] = *(const v2f*)(bSrc + (size_t)p * 4 * S4_H2);
        stage(0, aR, bR);
    }
    __syncthreads();

    int cur = 0;
    for (int kt = 0; kt < S4_H / 32; ++kt) {
        // issue global loads for tile kt+1 (overlaps WMMAs below)
        v8bf aN = {};
        v2f  bN[8];
        if (kt + 1 < S4_H / 32) {
            const __bf16* as2 = aSrc + (size_t)(kt + 1) * aStep;
            const float*  bs2 = bSrc + (size_t)(kt + 1) * bStep;
            aN = *(const v8bf*)as2;
#pragma unroll
            for (int p = 0; p < 8; ++p) bN[p] = *(const v2f*)(bs2 + (size_t)p * 4 * S4_H2);
        }

        // --- fragments + 4 WMMAs per wave from buffer `cur` ---
        v16bf af;
        {
            const __bf16* ap = &As[cur][(msub * 16 + ln15) * 40 + lh * 16];
            v8bf lo = *(const v8bf*)ap;
            v8bf hi = *(const v8bf*)(ap + 8);
            af = __builtin_shufflevector(lo, hi, 0, 1, 2, 3, 4, 5, 6, 7,
                                                 8, 9, 10, 11, 12, 13, 14, 15);
        }
#pragma unroll
        for (int nt = 0; nt < 4; ++nt) {
            const __bf16* bp = &Bs[cur][(hsel * 64 + nt * 16 + ln15) * 40 + lh * 16];
            v8bf lo = *(const v8bf*)bp;
            v8bf hi = *(const v8bf*)(bp + 8);
            v16bf bf = __builtin_shufflevector(lo, hi, 0, 1, 2, 3, 4, 5, 6, 7,
                                                       8, 9, 10, 11, 12, 13, 14, 15);
            acc[nt] = __builtin_amdgcn_wmma_f32_16x16x32_bf16(
                false, af, false, bf, (short)0, acc[nt], false, false);
        }

        if (kt + 1 < S4_H / 32) stage(cur ^ 1, aN, bN);
        __syncthreads();
        cur ^= 1;
    }

    // --- epilogue: stage Z tiles, then fused bias + GLU, coalesced stores ---
#pragma unroll
    for (int nt = 0; nt < 4; ++nt) {
#pragma unroll
        for (int r = 0; r < 8; ++r) {
            int mrow = msub * 16 + r + lh * 8;
            int ncol = hsel * 64 + nt * 16 + ln15;
            Zs[mrow * 132 + ncol] = acc[nt][r];
        }
    }
    __syncthreads();
#pragma unroll
    for (int pass = 0; pass < 4; ++pass) {
        int ml = (t >> 4) + pass * 16;
        int nl = (t & 15) * 4;
        v4f av  = *(const v4f*)&Zs[ml * 132 + nl];
        v4f gv  = *(const v4f*)&Zs[ml * 132 + 64 + nl];
        v4f ba  = *(const v4f*)&bout[n0 + nl];
        v4f bg  = *(const v4f*)&bout[S4_H + n0 + nl];
        v4f o;
        o.x = (av.x + ba.x) * s4_sigmoid(gv.x + bg.x);
        o.y = (av.y + ba.y) * s4_sigmoid(gv.y + bg.y);
        o.z = (av.z + ba.z) * s4_sigmoid(gv.z + bg.z);
        o.w = (av.w + ba.w) * s4_sigmoid(gv.w + bg.w);
        *(v4f*)&out[(m0 + ml) * S4_H + n0 + nl] = o;
    }
}

// ---------------------------------------------------------------------------
extern "C" void kernel_launch(void* const* d_in, const int* in_sizes, int n_in,
                              void* d_out, int out_size, void* d_ws, size_t ws_size,
                              hipStream_t stream) {
    const float* x      = (const float*)d_in[0];
    const float* log_dt = (const float*)d_in[1];
    const float* arl    = (const float*)d_in[2];
    const float* aim    = (const float*)d_in[3];
    const float* cre    = (const float*)d_in[4];
    const float* cim    = (const float*)d_in[5];
    const float* Dp     = (const float*)d_in[6];
    const float* W      = (const float*)d_in[7];
    const float* bout   = (const float*)d_in[8];
    float* out = (float*)d_out;

    char*   ws   = (char*)d_ws;
    float*  xt   = (float*)(ws);                                  // 64 MB
    float*  kern = (float*)(ws + ((size_t)64 << 20));             //  4 MB
    __bf16* yt   = (__bf16*)(ws + ((size_t)68 << 20));            // 32 MB

    s4_kern_gen<<<S4_H, 256, 0, stream>>>(log_dt, arl, aim, cre, cim, kern);
    s4_transpose<<<dim3(S4_H / 32, S4_L / 32, S4_B), dim3(32, 8), 0, stream>>>(x, xt);
    s4_conv_wmma<<<S4_H, 256, 0, stream>>>(xt, kern, Dp, yt);
    s4_gemm_glu<<<dim3(S4_B * S4_L / 64, S4_H / 64), 256, 0, stream>>>(yt, W, bout, out);
}
